// GraphMemoryVQ_EMA_24902220382584
// MI455X (gfx1250) — matrix operand order repaired and checked by reading, not compiled, not run on hardware
//
#include <hip/hip_runtime.h>
#include <hip/hip_bf16.h>
#include <cfloat>

// ---------------------------------------------------------------------------
// GraphMemoryVQ_EMA for gfx1250 (MI455X, wave32, WMMA)
//
// Kernels:
//   k_prep_codebook : codebook f32 -> bf16 hi/lo planes (ws) + ||c_j||^2
//   k_init          : d_out EMA buffers = DECAY*inputs, zero accumulators
//   k_argmin        : fused (-2 z@cT + ||c||^2 - 0.7*sigmoid(adj[prev]))
//                     + argmin via v_wmma_f32_16x16x32_bf16 (bf16x3 split)
//   k_scatter       : z_q gather, z_complex out, loss sumsq, EMA atomics
//   k_total         : n_total reduction + final loss scalar
//   k_codebook      : new_codebook = new_embed_avg / cs
// ---------------------------------------------------------------------------

typedef __attribute__((ext_vector_type(16))) __bf16    v16bf;
typedef __attribute__((ext_vector_type(8)))  float     v8f;
typedef __attribute__((ext_vector_type(4)))  unsigned  uvec4;
typedef __attribute__((ext_vector_type(4)))  float     fvec4;

#define NTOK   32768
#define LATENT 512
#define EMBED  1024
#define NSYM   8192
#define DECAY  0.995f
#define OMD    0.005f      // 1 - DECAY
#define GBS    0.7f
#define EPSV   1e-6f

// workspace layout (bytes)
#define WS_CHI   ((size_t)0)           // bf16 hi plane of codebook: 8192*1024*2
#define WS_CLO   ((size_t)16777216)    // bf16 lo plane
#define WS_CNORM ((size_t)33554432)    // 8192 f32
#define WS_IDX   ((size_t)33587200)    // 32768 int
#define WS_LOSS  ((size_t)33718272)    // f32 accumulator
#define WS_NTOT  ((size_t)33718276)    // f32

// output layout (floats), reference tuple order
static constexpr size_t OUT_Z    = 0;          // 32768*512*2 interleaved (re,im)
static constexpr size_t OUT_LOSS = 33554432;
static constexpr size_t OUT_IDX  = 33554433;   // 32768 (as float)
static constexpr size_t OUT_CB   = 33587201;   // 8192*1024
static constexpr size_t OUT_CS   = 41975809;   // 8192
static constexpr size_t OUT_EA   = 41984001;   // 8192*1024

__device__ __forceinline__ void split_bf16(float x, unsigned short& hi, unsigned short& lo) {
    unsigned u  = __float_as_uint(x);
    unsigned uh = (u + 0x8000u) & 0xFFFF0000u;     // round-half-up to bf16
    float    fh = __uint_as_float(uh);
    float    fl = x - fh;                           // residual, |fl| tiny
    hi = (unsigned short)(uh >> 16);
    lo = (unsigned short)(__float_as_uint(fl) >> 16);
}

union FragBF { v16bf v; uvec4 q[2]; };

__device__ __forceinline__ v8f wmma_bf16(v16bf a, v16bf b, v8f c) {
    return __builtin_amdgcn_wmma_f32_16x16x32_bf16(
        /*neg_a=*/false, a, /*neg_b=*/false, b,
        /*c_mod=*/(short)0, c, /*reuse_a=*/false, /*reuse_b=*/false);
}

// --------------------------------------------------------------------------
__global__ void k_prep_codebook(const float* __restrict__ cb,
                                unsigned short* __restrict__ chi,
                                unsigned short* __restrict__ clo,
                                float* __restrict__ cnorm) {
    const int j = blockIdx.x, t = threadIdx.x;
    float s = 0.f;
    for (int k = t; k < EMBED; k += 256) {
        float x = cb[(size_t)j * EMBED + k];
        unsigned short h, l; split_bf16(x, h, l);
        chi[(size_t)j * EMBED + k] = h;
        clo[(size_t)j * EMBED + k] = l;
        s += x * x;
    }
    __shared__ float red[256];
    red[t] = s; __syncthreads();
    for (int o = 128; o > 0; o >>= 1) { if (t < o) red[t] += red[t + o]; __syncthreads(); }
    if (t == 0) cnorm[j] = red[0];
}

// --------------------------------------------------------------------------
__global__ void k_init(const float* __restrict__ cs_in,
                       const float* __restrict__ ea_in,
                       float* __restrict__ out,
                       float* __restrict__ lossAcc,
                       float* __restrict__ ntot) {
    const size_t p = (size_t)blockIdx.x * 256 + threadIdx.x;   // 0 .. 8388607
    out[OUT_EA + p] = ea_in[p] * DECAY;
    if (p < NSYM) out[OUT_CS + p] = cs_in[p] * DECAY;
    if (p == 0) { *lossAcc = 0.f; *ntot = 0.f; }
}

// --------------------------------------------------------------------------
#define MWG    64
#define LDSROW 1032   // ushorts per token row (padded vs 1024 to break banks)

__global__ void __launch_bounds__(128, 1)
k_argmin(const float* __restrict__ zr, const float* __restrict__ zi,
         const int* __restrict__ prev, const float* __restrict__ adj,
         const unsigned short* __restrict__ chi,
         const unsigned short* __restrict__ clo,
         const float* __restrict__ cnorm,
         int* __restrict__ idxOut, float* __restrict__ out) {
    extern __shared__ unsigned short smem[];
    unsigned short* zhi = smem;
    unsigned short* zlo = smem + MWG * LDSROW;
    int* prevLds = (int*)(smem + 2 * MWG * LDSROW);

    const int tid  = threadIdx.x;
    const int tok0 = blockIdx.x * MWG;

    // stage z block (split to bf16 hi/lo) into LDS; z_flat = [z_real | z_imag]
    for (int i4 = tid; i4 < MWG * (EMBED / 4); i4 += 128) {
        const int row = i4 >> 8;              // EMBED/4 == 256
        const int k4  = (i4 & 255) * 4;
        const float* src = (k4 < LATENT)
            ? (zr + (size_t)(tok0 + row) * LATENT + k4)
            : (zi + (size_t)(tok0 + row) * LATENT + (k4 - LATENT));
        fvec4 v = *(const fvec4*)src;
#pragma unroll
        for (int e = 0; e < 4; ++e) {
            unsigned short h, l; split_bf16(v[e], h, l);
            zhi[row * LDSROW + k4 + e] = h;
            zlo[row * LDSROW + k4 + e] = l;
        }
    }
    if (tid < MWG) prevLds[tid] = prev[tok0 + tid];
    __syncthreads();

    const int wave = tid >> 5, lane = tid & 31;
    const int colq = lane & 15, hlf = lane >> 4;   // C: N=colq, M=r+8*hlf

    float bv[4][8]; int bi[4][8];
#pragma unroll
    for (int a = 0; a < 4; ++a)
#pragma unroll
        for (int r = 0; r < 8; ++r) { bv[a][r] = FLT_MAX; bi[a][r] = 0x7FFFFFFF; }

    for (int nt = wave; nt < NSYM / 16; nt += 4) {
        const int j0 = nt * 16;
        const int j  = j0 + colq;
        const float cn = cnorm[j];
        const unsigned short* bh0 = chi + (size_t)j * EMBED + hlf * 8;
        const unsigned short* bl0 = clo + (size_t)j * EMBED + hlf * 8;
        if (nt + 4 < NSYM / 16)
            __builtin_prefetch(chi + (size_t)(j + 64) * EMBED, 0, 0);

        v8f acc[4];
#pragma unroll
        for (int a = 0; a < 4; ++a)
#pragma unroll
            for (int r = 0; r < 8; ++r) acc[a][r] = 0.f;

#pragma unroll 1
        for (int kk = 0; kk < EMBED; kk += 32) {
            FragBF bh, bl;
            bh.q[0] = *(const uvec4*)(bh0 + kk);
            bh.q[1] = *(const uvec4*)(bh0 + kk + 16);
            bl.q[0] = *(const uvec4*)(bl0 + kk);
            bl.q[1] = *(const uvec4*)(bl0 + kk + 16);
#pragma unroll
            for (int a = 0; a < 4; ++a) {
                const unsigned short* pah = zhi + (a * 16 + colq) * LDSROW + kk + hlf * 8;
                const unsigned short* pal = zlo + (a * 16 + colq) * LDSROW + kk + hlf * 8;
                FragBF ah, al;
                ah.q[0] = *(const uvec4*)(pah);
                ah.q[1] = *(const uvec4*)(pah + 16);
                al.q[0] = *(const uvec4*)(pal);
                al.q[1] = *(const uvec4*)(pal + 16);
                acc[a] = wmma_bf16(ah.v, bh.v, acc[a]);   // hi*hi
                acc[a] = wmma_bf16(ah.v, bl.v, acc[a]);   // hi*lo
                acc[a] = wmma_bf16(al.v, bh.v, acc[a]);   // lo*hi
            }
        }

        // epilogue: d = ||c||^2 - 2 z.c - 0.7*sigmoid(adj[prev][j]); running argmin
#pragma unroll
        for (int a = 0; a < 4; ++a)
#pragma unroll
            for (int r = 0; r < 8; ++r) {
                const int lrow = a * 16 + hlf * 8 + r;
                const int p    = prevLds[lrow];
                const float ad = adj[(size_t)p * NSYM + j];
                const float bias = GBS / (1.f + __expf(-ad));
                const float s = cn - 2.f * acc[a][r] - bias;
                if (s < bv[a][r]) { bv[a][r] = s; bi[a][r] = j; }
            }
    }

    // reduce across the 16 lanes sharing each row (first-index tie-break)
#pragma unroll
    for (int a = 0; a < 4; ++a)
#pragma unroll
        for (int r = 0; r < 8; ++r) {
            float v = bv[a][r]; int ix = bi[a][r];
            for (int m = 8; m >= 1; m >>= 1) {
                float ov = __shfl_xor(v, m, 16);
                int   oi = __shfl_xor(ix, m, 16);
                if (ov < v || (ov == v && oi < ix)) { v = ov; ix = oi; }
            }
            bv[a][r] = v; bi[a][r] = ix;
        }

    __syncthreads();                 // compute done; safe to overlay z-LDS
    float* rv = (float*)smem;        // [4][MWG]
    int*   ri = (int*)(rv + 4 * MWG);
    if (colq == 0) {
#pragma unroll
        for (int a = 0; a < 4; ++a)
#pragma unroll
            for (int r = 0; r < 8; ++r) {
                const int lrow = a * 16 + hlf * 8 + r;
                rv[wave * MWG + lrow] = bv[a][r];
                ri[wave * MWG + lrow] = bi[a][r];
            }
    }
    __syncthreads();
    if (tid < MWG) {
        float v = rv[tid]; int ix = ri[tid];
        for (int w = 1; w < 4; ++w) {
            float ov = rv[w * MWG + tid]; int oi = ri[w * MWG + tid];
            if (ov < v || (ov == v && oi < ix)) { v = ov; ix = oi; }
        }
        idxOut[tok0 + tid] = ix;
        out[OUT_IDX + tok0 + tid] = (float)ix;
    }
}

// --------------------------------------------------------------------------
__global__ void k_scatter(const float* __restrict__ zr, const float* __restrict__ zi,
                          const float* __restrict__ cb, const int* __restrict__ idx,
                          float* __restrict__ out, float* __restrict__ lossAcc) {
    const int tok = blockIdx.x, t = threadIdx.x;
    const int j = idx[tok];
    float local = 0.f;
    for (int k = t; k < EMBED; k += 256) {
        const float zq = cb[(size_t)j * EMBED + k];          // pre-EMA codebook
        const float z  = (k < LATENT) ? zr[(size_t)tok * LATENT + k]
                                      : zi[(size_t)tok * LATENT + k - LATENT];
        const float d = zq - z;
        local += d * d;
        // straight-through forward == z_q; complex interleaved (re, im)
        if (k < LATENT) out[OUT_Z + ((size_t)tok * LATENT + k) * 2]                = zq;
        else            out[OUT_Z + ((size_t)tok * LATENT + (k - LATENT)) * 2 + 1] = zq;
        atomicAdd(&out[OUT_EA + (size_t)j * EMBED + k], OMD * z);   // segment_sum of z_flat
    }
    __shared__ float red[256];
    red[t] = local; __syncthreads();
    for (int o = 128; o > 0; o >>= 1) { if (t < o) red[t] += red[t + o]; __syncthreads(); }
    if (t == 0) {
        atomicAdd(lossAcc, red[0]);
        atomicAdd(&out[OUT_CS + j], OMD);                    // counts*(1-DECAY)
    }
}

// --------------------------------------------------------------------------
__global__ void k_total(const float* __restrict__ out_cs, float* __restrict__ ntot,
                        const float* __restrict__ lossAcc, float* __restrict__ out) {
    __shared__ float red[256];
    const int t = threadIdx.x;
    float s = 0.f;
    for (int i = t; i < NSYM; i += 256) s += out_cs[i];
    red[t] = s; __syncthreads();
    for (int o = 128; o > 0; o >>= 1) { if (t < o) red[t] += red[t + o]; __syncthreads(); }
    if (t == 0) {
        *ntot = red[0];
        // loss_vq + 0.25*loss_commit, forward-identical terms
        out[OUT_LOSS] = 1.25f * (*lossAcc) / ((float)NTOK * (float)EMBED);
    }
}

// --------------------------------------------------------------------------
__global__ void k_codebook(const float* __restrict__ ntotp, float* __restrict__ out) {
    const int j = blockIdx.x, t = threadIdx.x;
    const float nt_ = *ntotp;
    const float ncs = out[OUT_CS + j];
    const float cs  = (ncs + EPSV) / (nt_ + (float)NSYM * EPSV) * nt_;
    const float inv = 1.f / cs;
    for (int k = t; k < EMBED; k += 256)
        out[OUT_CB + (size_t)j * EMBED + k] = out[OUT_EA + (size_t)j * EMBED + k] * inv;
}

// --------------------------------------------------------------------------
extern "C" void kernel_launch(void* const* d_in, const int* in_sizes, int n_in,
                              void* d_out, int out_size, void* d_ws, size_t ws_size,
                              hipStream_t stream) {
    const float* zr   = (const float*)d_in[0];
    const float* zi   = (const float*)d_in[1];
    const int*   prev = (const int*)d_in[2];
    const float* cb   = (const float*)d_in[3];
    const float* cs   = (const float*)d_in[4];
    const float* ea   = (const float*)d_in[5];
    const float* adj  = (const float*)d_in[6];
    float* out = (float*)d_out;
    char*  ws  = (char*)d_ws;

    unsigned short* chi   = (unsigned short*)(ws + WS_CHI);
    unsigned short* clo   = (unsigned short*)(ws + WS_CLO);
    float* cnorm          = (float*)(ws + WS_CNORM);
    int*   midx           = (int*)(ws + WS_IDX);
    float* lossAcc        = (float*)(ws + WS_LOSS);
    float* ntot           = (float*)(ws + WS_NTOT);

    k_prep_codebook<<<NSYM, 256, 0, stream>>>(cb, chi, clo, cnorm);
    k_init<<<(NSYM * EMBED) / 256, 256, 0, stream>>>(cs, ea, out, lossAcc, ntot);

    const size_t smem = (size_t)2 * MWG * LDSROW * sizeof(unsigned short)
                        + MWG * sizeof(int);   // ~258.4 KB, fits 320 KB WGP LDS
    hipFuncSetAttribute((const void*)k_argmin,
                        hipFuncAttributeMaxDynamicSharedMemorySize, (int)smem);
    k_argmin<<<NTOK / MWG, 128, smem, stream>>>(zr, zi, prev, adj, chi, clo,
                                                cnorm, midx, out);

    k_scatter<<<NTOK, 256, 0, stream>>>(zr, zi, cb, midx, out, lossAcc);
    k_total<<<1, 256, 0, stream>>>(out + OUT_CS, ntot, lossAcc, out);
    k_codebook<<<NSYM, 256, 0, stream>>>(ntot, out);
}
